// MultimodalGNN_17068200034897
// MI455X (gfx1250) — matrix operand order repaired and checked by reading, not compiled, not run on hardware
//
#include <hip/hip_runtime.h>

typedef __attribute__((ext_vector_type(16))) __bf16 v16bf;
typedef __attribute__((ext_vector_type(8)))  float  v8f;

// ---------- helpers ----------

// Pack two f32 into one dword of two bf16 (RNE) using CDNA5 mixed-precision FMA
// converts: v_fma_mixlo_bf16 / v_fma_mixhi_bf16 (VOP3P opc 62/63).
__device__ __forceinline__ unsigned int f2bf_pk(float lo, float hi) {
    unsigned int d;
    asm("v_fma_mixlo_bf16 %0, %1, 1.0, 0\n\t"
        "v_fma_mixhi_bf16 %0, %2, 1.0, 0"
        : "=v"(d)
        : "v"(lo), "v"(hi));
    return d;
}

union FragBF {
    v16bf v;
    unsigned int ui[8];
    uint4 q[2];
};

// Native no-return FP32 atomic add (GLOBAL_ATOMIC_ADD_F32, tracked by STOREcnt;
// S_ENDPGM performs an implicit wait-idle). Avoids any CAS-loop expansion.
__device__ __forceinline__ void atomic_add_f32(float* p, float v) {
    asm volatile("global_atomic_add_f32 %0, %1, off"
                 :
                 : "v"(p), "v"(v)
                 : "memory");
}

// ---------- degree / norm kernels ----------

__global__ __launch_bounds__(256) void fill_k(float* __restrict__ p, int n, float v) {
    int i = blockIdx.x * 256 + threadIdx.x;
    if (i < n) p[i] = v;
}

__global__ __launch_bounds__(256) void deg_count_k(float* deg, const int* __restrict__ dst, int E) {
    int e = blockIdx.x * 256 + threadIdx.x;
    if (e < E) atomic_add_f32(deg + dst[e], 1.0f);
}

__global__ __launch_bounds__(256) void rsqrt_k(float* __restrict__ p, int n) {
    int i = blockIdx.x * 256 + threadIdx.x;
    if (i < n) {
        float d = p[i];
        p[i] = (d > 0.0f) ? rsqrtf(d) : 0.0f;
    }
}

__global__ __launch_bounds__(256) void norm_k(float* __restrict__ nrm,
                                              const float* __restrict__ dinv,
                                              const int* __restrict__ src,
                                              const int* __restrict__ dst, int E) {
    int e = blockIdx.x * 256 + threadIdx.x;
    if (e < E) nrm[e] = dinv[src[e]] * dinv[dst[e]];
}

// ---------- WMMA GEMM: Y[n,NOUT] = X[n,K] @ W[K,NOUT] (bf16 compute, f32 acc) ----------
// Block = 256 threads = 8 waves; each wave computes a 16 x NOUT strip.

template<int K, int NOUT>
__global__ __launch_bounds__(256) void gcn_gemm_wmma(const float* __restrict__ X,
                                                     const float* __restrict__ W,
                                                     float* __restrict__ Y, int nrows) {
    constexpr int KT  = K / 32;            // k-steps of 32
    constexpr int NT  = NOUT / 16;         // 16-wide n tiles
    constexpr int KTC = (KT < 8) ? KT : 8; // k-steps staged in LDS per chunk

    // B fragments in LDS: per (ktl, nt, lane): 16 bf16 = 8 dwords
    __shared__ unsigned int ldsB[KTC * NT * 32 * 8];

    const int tid  = threadIdx.x;
    const int wave = tid >> 5;
    const int lane = tid & 31;
    const int half = lane >> 4;

    const int rowBase = blockIdx.x * 128 + wave * 16;
    int r  = rowBase + (lane & 15);
    int rc = (r < nrows) ? r : (nrows - 1);
    const float* xrow = X + (size_t)rc * K;

    v8f zero = {};
    v8f acc[NT];
#pragma unroll
    for (int nt = 0; nt < NT; ++nt) acc[nt] = zero;

    for (int ktc = 0; ktc < KT; ktc += KTC) {
        __syncthreads();
        // Stage weight fragments for this k chunk (B layout: element i -> K = 8*half + (i&7) + 16*(i>=8)).
        for (int u = tid; u < KTC * NT * 32 * 8; u += 256) {
            int fragLane = u >> 3;
            int ii  = (u & 7) << 1;          // even bf16 element index
            int l   = fragLane & 31;
            int fr  = fragLane >> 5;
            int nt  = fr % NT;
            int ktl = fr / NT;
            int k   = (ktc + ktl) * 32 + (l >> 4) * 8 + (ii & 7) + (ii >> 3) * 16;
            int n   = nt * 16 + (l & 15);
            ldsB[u] = f2bf_pk(W[(size_t)k * NOUT + n], W[(size_t)(k + 1) * NOUT + n]);
        }
        __syncthreads();

#pragma unroll
        for (int ktl = 0; ktl < KTC; ++ktl) {
            const int kt = ktc + ktl;
            // A fragment: lane holds row (lane&15); element i -> K = 8*half + (i&7) + 16*(i>=8)
            const float* xp = xrow + kt * 32 + half * 8;
            float4 f0 = *(const float4*)(xp + 0);
            float4 f1 = *(const float4*)(xp + 4);
            float4 f2 = *(const float4*)(xp + 16);
            float4 f3 = *(const float4*)(xp + 20);
            FragBF a;
            a.ui[0] = f2bf_pk(f0.x, f0.y);
            a.ui[1] = f2bf_pk(f0.z, f0.w);
            a.ui[2] = f2bf_pk(f1.x, f1.y);
            a.ui[3] = f2bf_pk(f1.z, f1.w);
            a.ui[4] = f2bf_pk(f2.x, f2.y);
            a.ui[5] = f2bf_pk(f2.z, f2.w);
            a.ui[6] = f2bf_pk(f3.x, f3.y);
            a.ui[7] = f2bf_pk(f3.z, f3.w);

#pragma unroll
            for (int nt = 0; nt < NT; ++nt) {
                FragBF b;
                int base = ((ktl * NT + nt) * 32 + lane) * 8;
                b.q[0] = *(const uint4*)(&ldsB[base]);
                b.q[1] = *(const uint4*)(&ldsB[base + 4]);
                acc[nt] = __builtin_amdgcn_wmma_f32_16x16x32_bf16(
                    false, a.v, false, b.v, (short)0, acc[nt], false, false);
            }
        }
    }

    // C/D layout: element j -> row M = 8*half + j; column = nt*16 + (lane&15)
#pragma unroll
    for (int nt = 0; nt < NT; ++nt) {
#pragma unroll
        for (int j = 0; j < 8; ++j) {
            int gr = rowBase + half * 8 + j;
            if (gr < nrows)
                Y[(size_t)gr * NOUT + nt * 16 + (lane & 15)] = acc[nt][j];
        }
    }
}

// ---------- edge scatter-add: out[dst] += h[src] * norm  ----------

template<int F>
__global__ __launch_bounds__(256) void scatter_edges(const float* __restrict__ h,
                                                     const float* __restrict__ nrm,
                                                     const int* __restrict__ src,
                                                     const int* __restrict__ dst,
                                                     float* out, int E) {
    size_t t = (size_t)blockIdx.x * 256 + threadIdx.x;
    size_t e = t / F;
    int    f = (int)(t & (F - 1));
    if (e >= (size_t)E) return;
    int s = src[e];
    int d = dst[e];
    float v = h[(size_t)s * F + f] * nrm[e];
    atomic_add_f32(out + (size_t)d * F + f, v);
}

// ---------- finalize: out += h * dinv^2 (self loop) + bias, optional ReLU ----------

template<int F, bool RELU>
__global__ __launch_bounds__(256) void finalize_k(float* out,
                                                  const float* __restrict__ h,
                                                  const float* __restrict__ dinv,
                                                  const float* __restrict__ bias, int n) {
    size_t t = (size_t)blockIdx.x * 256 + threadIdx.x;
    if (t >= (size_t)n * F) return;
    size_t i = t / F;
    int    f = (int)(t & (F - 1));
    float di = dinv[i];
    float v = out[t] + h[t] * di * di + bias[f];
    out[t] = RELU ? fmaxf(v, 0.0f) : v;
}

// ---------- launch ----------

extern "C" void kernel_launch(void* const* d_in, const int* in_sizes, int n_in,
                              void* d_out, int out_size, void* d_ws, size_t ws_size,
                              hipStream_t stream) {
    const float* x  = (const float*)d_in[0];   // [N,768]
    const int*   ei = (const int*)d_in[1];     // [2,E]
    const float* W1 = (const float*)d_in[2];   // [768,64]
    const float* b1 = (const float*)d_in[3];   // [64]
    const float* W2 = (const float*)d_in[4];   // [64,32]
    const float* b2 = (const float*)d_in[5];   // [32]

    const int N = in_sizes[0] / 768;
    const int E = in_sizes[1] / 2;
    const int* src = ei;
    const int* dst = ei + E;

    // workspace layout (f32): dinv[N] | norm[E] | h[N*64] (layer1 h, reused as layer2 h) | out1[N*64]
    float* dinv  = (float*)d_ws;
    float* normv = dinv + N;
    float* h     = normv + E;
    float* out1  = h + (size_t)N * 64;
    float* outF  = (float*)d_out;

    const int bN = (N + 255) / 256;
    const int bE = (E + 255) / 256;
    const int gB = (N + 127) / 128;

    // degrees (self-loop contributes 1), dinv, per-edge norm
    fill_k<<<bN, 256, 0, stream>>>(dinv, N, 1.0f);
    deg_count_k<<<bE, 256, 0, stream>>>(dinv, dst, E);
    rsqrt_k<<<bN, 256, 0, stream>>>(dinv, N);
    norm_k<<<bE, 256, 0, stream>>>(normv, dinv, src, dst, E);

    // layer 1: h = x @ W1 ; out1 = relu(scatter(h*norm) + h*dinv^2 + b1)
    gcn_gemm_wmma<768, 64><<<gB, 256, 0, stream>>>(x, W1, h, N);
    hipMemsetAsync(out1, 0, (size_t)N * 64 * sizeof(float), stream);
    {
        size_t tot = (size_t)E * 64;
        scatter_edges<64><<<(int)((tot + 255) / 256), 256, 0, stream>>>(h, normv, src, dst, out1, E);
    }
    {
        size_t tot = (size_t)N * 64;
        finalize_k<64, true><<<(int)((tot + 255) / 256), 256, 0, stream>>>(out1, h, dinv, b1, N);
    }

    // layer 2: h2 = out1 @ W2 ; out = scatter(h2*norm) + h2*dinv^2 + b2
    gcn_gemm_wmma<64, 32><<<gB, 256, 0, stream>>>(out1, W2, h, N);
    hipMemsetAsync(d_out, 0, (size_t)N * 32 * sizeof(float), stream);
    {
        size_t tot = (size_t)E * 32;
        scatter_edges<32><<<(int)((tot + 255) / 256), 256, 0, stream>>>(h, normv, src, dst, outF, E);
    }
    {
        size_t tot = (size_t)N * 32;
        finalize_k<32, false><<<(int)((tot + 255) / 256), 256, 0, stream>>>(outF, h, dinv, b2, N);
    }
}
